// MultiHeadAttention_63247688401023
// MI455X (gfx1250) — matrix operand order repaired and checked by reading, not compiled, not run on hardware
//
#include <hip/hip_runtime.h>
#include <hip/hip_bf16.h>

// ---------------------------------------------------------------------------
// Multi-head attention on gfx1250 (MI455X), wave32 + v_wmma_f32_16x16x32_f16.
//   B=2, S=2048, D=1024, H=16, Dh=64
// Pipeline:
//   1) cvt_x / cvt_wT : fp32 -> f16 (weights transposed so WMMA B-operand
//      loads are contiguous 16B chunks per lane)
//   2) gemm: Q/K (head-split rowmajor f16), V (head-split transposed f16).
//      B operand double-buffered through LDS with GLOBAL_LOAD_ASYNC_TO_LDS
//      (ASYNCcnt + s_wait_asynccnt), 8 WMMAs per wave per k-step.
//   3) attn: flash-style streaming softmax(QK^T/8)V per 16-row q-tile,
//      WMMA scores + PV, ds_swizzle row reductions, LDS P transpose.
//   4) gemm: out = attn_out@wo + bo (fp32 to d_out)
// ---------------------------------------------------------------------------

typedef __attribute__((ext_vector_type(16))) _Float16 v16h;
typedef __attribute__((ext_vector_type(8)))  _Float16 v8h;
typedef __attribute__((ext_vector_type(8)))  float    v8f;

#define D_MODEL   1024
#define SEQ       2048
#define NHEADS    16
#define HEAD_DIM  64
#define BATCH     2
#define ROWS      (BATCH * SEQ)          // 4096

__device__ __forceinline__ v16h cat8(v8h lo, v8h hi) {
  return __builtin_shufflevector(lo, hi, 0,1,2,3,4,5,6,7,8,9,10,11,12,13,14,15);
}

// A-operand tile (16x32, f16, row-major source, leading-dim ld).
// lane l holds row m=l&15; halves e=0..15 -> K=(l>>4)*16+e (32B run).
__device__ __forceinline__ v16h load_a16(const _Float16* base, int ld, int m0,
                                         int k0, int lane) {
  int m  = m0 + (lane & 15);
  int hi = lane >> 4;
  const _Float16* p = base + (size_t)m * ld + k0 + hi * 16;
  return cat8(*(const v8h*)p, *(const v8h*)(p + 8));
}

// B-operand tile (32x16, f16) from column-major storage (bT[n][k] row-major).
// lane l holds column n=l&15; halves e=0..15 -> K=(l>>4)*16+e (32B run).
__device__ __forceinline__ v16h load_b16(const _Float16* baseT, int ld, int n0,
                                         int k0, int lane) {
  int n  = n0 + (lane & 15);
  int hi = lane >> 4;
  const _Float16* p = baseT + (size_t)n * ld + k0 + hi * 16;
  return cat8(*(const v8h*)p, *(const v8h*)(p + 8));
}

// wave32 cross-lane reductions inside each 16-lane half via ds_swizzle xor.
template <int OFF>
__device__ __forceinline__ float swz(float x) {
  return __int_as_float(__builtin_amdgcn_ds_swizzle(__float_as_int(x), OFF));
}
__device__ __forceinline__ float hmax16(float x) {
  x = fmaxf(x, swz<0x1f | (1 << 10)>(x));
  x = fmaxf(x, swz<0x1f | (2 << 10)>(x));
  x = fmaxf(x, swz<0x1f | (4 << 10)>(x));
  x = fmaxf(x, swz<0x1f | (8 << 10)>(x));
  return x;
}
__device__ __forceinline__ float hsum16(float x) {
  x += swz<0x1f | (1 << 10)>(x);
  x += swz<0x1f | (2 << 10)>(x);
  x += swz<0x1f | (4 << 10)>(x);
  x += swz<0x1f | (8 << 10)>(x);
  return x;
}

// --------------------------- conversion kernels ----------------------------

__global__ void cvt_x_kernel(const float* __restrict__ x,
                             _Float16* __restrict__ xh, int n) {
  int i = blockIdx.x * 256 + threadIdx.x;
  if (i < n) xh[i] = (_Float16)x[i];
}

// wT[n*1024 + k] = (f16) w[k*1024 + n]
__global__ void cvt_wT_kernel(const float* __restrict__ w,
                              _Float16* __restrict__ wT) {
  int i = blockIdx.x * 256 + threadIdx.x;   // over 1M elements
  int n = i >> 10;
  int k = i & 1023;
  wT[i] = (_Float16)w[k * D_MODEL + n];
}

// ------------------------------- GEMM --------------------------------------
// C[4096,1024] = A[4096,1024] @ B[1024,1024]; A rowmajor f16, B as BT[n][k].
// Block tile 128x64 (4 waves x 32 rows); per wave 2 M-subtiles x 4 N-subtiles
// = 8 accumulators. B slab (64 cols x 32 k = 4KB) is shared by all 4 waves
// and double-buffered in LDS via global_load_async_to_lds_b128.
// mode 0: f16 head-split rowmajor  (Q/K)
// mode 1: f16 head-split transposed (V)
// mode 2: f32 rowmajor + bias to d_out
__global__ __launch_bounds__(128) void gemm_kernel(
    const _Float16* __restrict__ A, const _Float16* __restrict__ BT,
    const float* __restrict__ bias, void* __restrict__ out, int mode) {
  __shared__ __align__(16) _Float16 bstage[2][64 * 32];   // 2 x 4KB

  int tid  = threadIdx.x;
  int lane = tid & 31;
  int wave = tid >> 5;
  int m0 = blockIdx.x * 128 + wave * 32;
  int n0 = blockIdx.y * 64;

  // Async-copy mapping: 256 16-byte chunks cover the 64x32 slab; each of the
  // 128 threads moves chunks (tid) and (tid+128).
  int row0 = tid >> 2,           q0 = tid & 3;          // chunk tid
  int row1 = (tid + 128) >> 2,   q1 = (tid + 128) & 3;  // chunk tid+128
  const _Float16* g0base = BT + (size_t)(n0 + row0) * D_MODEL + q0 * 8;
  const _Float16* g1base = BT + (size_t)(n0 + row1) * D_MODEL + q1 * 8;
  unsigned l0a = (unsigned)(uintptr_t)&bstage[0][row0 * 32 + q0 * 8];
  unsigned l1a = (unsigned)(uintptr_t)&bstage[0][row1 * 32 + q1 * 8];
  unsigned l0b = (unsigned)(uintptr_t)&bstage[1][row0 * 32 + q0 * 8];
  unsigned l1b = (unsigned)(uintptr_t)&bstage[1][row1 * 32 + q1 * 8];

#define ISSUE_COPY(buf, k)                                                    \
  do {                                                                        \
    unsigned long long ga = (unsigned long long)(uintptr_t)(g0base + (k));    \
    unsigned long long gb = (unsigned long long)(uintptr_t)(g1base + (k));    \
    unsigned la = (buf) ? l0b : l0a;                                          \
    unsigned lb = (buf) ? l1b : l1a;                                          \
    asm volatile("global_load_async_to_lds_b128 %0, %1, off"                  \
                 :: "v"(la), "v"(ga) : "memory");                             \
    asm volatile("global_load_async_to_lds_b128 %0, %1, off"                  \
                 :: "v"(lb), "v"(gb) : "memory");                             \
  } while (0)

  v8f acc[8] = {};

  ISSUE_COPY(0, 0);
  for (int kt = 0; kt < D_MODEL / 32; ++kt) {
    int k = kt * 32;
    int cur = kt & 1;
    if (kt + 1 < D_MODEL / 32) {
      ISSUE_COPY(cur ^ 1, k + 32);
      // async loads retire in order: <=2 outstanding leaves only the two
      // next-buffer copies in flight, current buffer is complete.
      asm volatile("s_wait_asynccnt 0x2" ::: "memory");
    } else {
      asm volatile("s_wait_asynccnt 0x0" ::: "memory");
    }
    __syncthreads();   // all waves' slab writes for buf[cur] visible

    v16h a0 = load_a16(A, D_MODEL, m0,      k, lane);
    v16h a1 = load_a16(A, D_MODEL, m0 + 16, k, lane);
#pragma unroll
    for (int j = 0; j < 4; ++j) {
      const _Float16* p =
          &bstage[cur][(j * 16 + (lane & 15)) * 32 + (lane >> 4) * 16];
      v16h b = cat8(*(const v8h*)p, *(const v8h*)(p + 8));
      acc[j] = __builtin_amdgcn_wmma_f32_16x16x32_f16(
          false, a0, false, b, (short)0, acc[j], false, false);
      acc[4 + j] = __builtin_amdgcn_wmma_f32_16x16x32_f16(
          false, a1, false, b, (short)0, acc[4 + j], false, false);
    }
    __syncthreads();   // done reading buf[cur] before it is refilled
  }
#undef ISSUE_COPY

  int hi = lane >> 4;
  int nl = lane & 15;
#pragma unroll
  for (int i = 0; i < 2; ++i) {
#pragma unroll
    for (int j = 0; j < 4; ++j) {
      int c = n0 + j * 16 + nl;
      float bv = bias[c];
#pragma unroll
      for (int r = 0; r < 8; ++r) {
        int gr = m0 + i * 16 + hi * 8 + r;        // global row in [0,4096)
        float v = acc[i * 4 + j][r] + bv;
        if (mode == 2) {
          ((float*)out)[(size_t)gr * D_MODEL + c] = v;
        } else {
          int b = gr >> 11, s = gr & (SEQ - 1);
          int h = c >> 6,  d = c & (HEAD_DIM - 1);
          _Float16* o = (_Float16*)out;
          if (mode == 0)
            o[(((size_t)(b * NHEADS + h) * SEQ + s) * HEAD_DIM) + d] =
                (_Float16)v;
          else
            o[(((size_t)(b * NHEADS + h) * HEAD_DIM + d) * SEQ) + s] =
                (_Float16)v;
        }
      }
    }
  }
}

// ----------------------------- attention -----------------------------------
// One wave per (b,h, 16-row q-tile). Streams t in chunks of 32 with online
// softmax; accumulator O is 16x64 fp32 in C-layout (4 tiles of v8f).
__global__ __launch_bounds__(32) void attn_kernel(
    const _Float16* __restrict__ Qh, const _Float16* __restrict__ Kh,
    const _Float16* __restrict__ Vt, _Float16* __restrict__ aout) {
  __shared__ __align__(16) _Float16 pbuf[16 * 32];   // 1KB P transpose staging

  int lane = threadIdx.x & 31;
  int rt = blockIdx.x;                 // q row tile, 0..127
  int bh = blockIdx.y;                 // 0..31
  int b = bh >> 4, h = bh & 15;
  const _Float16* Q = Qh + (size_t)bh * SEQ * HEAD_DIM;
  const _Float16* K = Kh + (size_t)bh * SEQ * HEAD_DIM;
  const _Float16* V = Vt + (size_t)bh * HEAD_DIM * SEQ;
  int r0 = rt * 16;
  int hi = lane >> 4, nl = lane & 15;

  v16h qa0 = load_a16(Q, HEAD_DIM, r0, 0, lane);
  v16h qa1 = load_a16(Q, HEAD_DIM, r0, 32, lane);

  v8f o[4] = {};
  float mrow[8], lrow[8];
#pragma unroll
  for (int r = 0; r < 8; ++r) { mrow[r] = -1e30f; lrow[r] = 0.0f; }

  const float scale = 0.125f;          // 1/sqrt(64)

  for (int t0 = 0; t0 < SEQ; t0 += 32) {
    v8f s0 = {}, s1 = {};
    {
      v16h b00 = load_b16(K, HEAD_DIM, t0, 0, lane);
      v16h b01 = load_b16(K, HEAD_DIM, t0, 32, lane);
      s0 = __builtin_amdgcn_wmma_f32_16x16x32_f16(false, qa0, false, b00,
                                                  (short)0, s0, false, false);
      s0 = __builtin_amdgcn_wmma_f32_16x16x32_f16(false, qa1, false, b01,
                                                  (short)0, s0, false, false);
      v16h b10 = load_b16(K, HEAD_DIM, t0 + 16, 0, lane);
      v16h b11 = load_b16(K, HEAD_DIM, t0 + 16, 32, lane);
      s1 = __builtin_amdgcn_wmma_f32_16x16x32_f16(false, qa0, false, b10,
                                                  (short)0, s1, false, false);
      s1 = __builtin_amdgcn_wmma_f32_16x16x32_f16(false, qa1, false, b11,
                                                  (short)0, s1, false, false);
    }

    float p0[8], p1[8];
#pragma unroll
    for (int r = 0; r < 8; ++r) {
      float x0 = s0[r] * scale;
      float x1 = s1[r] * scale;
      float tmax = hmax16(fmaxf(x0, x1));
      float mnew = fmaxf(mrow[r], tmax);
      float alpha = __expf(mrow[r] - mnew);
      float e0 = __expf(x0 - mnew);
      float e1 = __expf(x1 - mnew);
      float rs = hsum16(e0 + e1);
      lrow[r] = lrow[r] * alpha + rs;
      mrow[r] = mnew;
      p0[r] = e0;
      p1[r] = e1;
      o[0][r] *= alpha; o[1][r] *= alpha; o[2][r] *= alpha; o[3][r] *= alpha;
    }

#pragma unroll
    for (int r = 0; r < 8; ++r) {
      int m = hi * 8 + r;
      pbuf[m * 32 + nl]      = (_Float16)p0[r];
      pbuf[m * 32 + 16 + nl] = (_Float16)p1[r];
    }
    __syncthreads();
    v16h pa;
    {
      const _Float16* pp = &pbuf[nl * 32 + hi * 16];
      pa = cat8(*(const v8h*)pp, *(const v8h*)(pp + 8));
    }
    __syncthreads();

#pragma unroll
    for (int j = 0; j < 4; ++j) {
      const _Float16* vp = V + (size_t)(j * 16 + nl) * SEQ + t0 + hi * 16;
      v16h vb = cat8(*(const v8h*)vp, *(const v8h*)(vp + 8));
      o[j] = __builtin_amdgcn_wmma_f32_16x16x32_f16(false, pa, false, vb,
                                                    (short)0, o[j], false, false);
    }
  }

#pragma unroll
  for (int j = 0; j < 4; ++j) {
    int c = h * HEAD_DIM + j * 16 + nl;
#pragma unroll
    for (int r = 0; r < 8; ++r) {
      int s = r0 + hi * 8 + r;
      float v = o[j][r] / lrow[r];
      aout[(size_t)(b * SEQ + s) * D_MODEL + c] = (_Float16)v;
    }
  }
}

// ----------------------------- launcher ------------------------------------

extern "C" void kernel_launch(void* const* d_in, const int* in_sizes, int n_in,
                              void* d_out, int out_size, void* d_ws, size_t ws_size,
                              hipStream_t stream) {
  const float* x  = (const float*)d_in[0];
  const float* wq = (const float*)d_in[1];
  const float* bq = (const float*)d_in[2];
  const float* wk = (const float*)d_in[3];
  const float* bk = (const float*)d_in[4];
  const float* wv = (const float*)d_in[5];
  const float* bv = (const float*)d_in[6];
  const float* wo = (const float*)d_in[7];
  const float* bo = (const float*)d_in[8];
  float* out = (float*)d_out;

  // workspace layout (f16 elements); ~40MB total
  _Float16* ws = (_Float16*)d_ws;
  const size_t MB1 = (size_t)1024 * 1024;
  _Float16* xh   = ws;                         // 4M halves (reused as attn_out)
  _Float16* wqT  = ws + 4 * MB1;               // 1M
  _Float16* wkT  = ws + 5 * MB1;
  _Float16* wvT  = ws + 6 * MB1;
  _Float16* woT  = ws + 7 * MB1;
  _Float16* Qh   = ws + 8 * MB1;               // 4M
  _Float16* Kh   = ws + 12 * MB1;              // 4M
  _Float16* Vt   = ws + 16 * MB1;              // 4M
  _Float16* aout = xh;                         // recycle xh after QKV GEMMs

  {
    int n = ROWS * D_MODEL;                    // 4M
    cvt_x_kernel<<<(n + 255) / 256, 256, 0, stream>>>(x, xh, n);
    cvt_wT_kernel<<<(D_MODEL * D_MODEL) / 256, 256, 0, stream>>>(wq, wqT);
    cvt_wT_kernel<<<(D_MODEL * D_MODEL) / 256, 256, 0, stream>>>(wk, wkT);
    cvt_wT_kernel<<<(D_MODEL * D_MODEL) / 256, 256, 0, stream>>>(wv, wvT);
    cvt_wT_kernel<<<(D_MODEL * D_MODEL) / 256, 256, 0, stream>>>(wo, woT);
  }

  dim3 ggrid(ROWS / 128, D_MODEL / 64);        // 32 x 16 blocks, 128 thr
  gemm_kernel<<<ggrid, 128, 0, stream>>>(xh, wqT, bq, Qh, 0);
  gemm_kernel<<<ggrid, 128, 0, stream>>>(xh, wkT, bk, Kh, 0);
  gemm_kernel<<<ggrid, 128, 0, stream>>>(xh, wvT, bv, Vt, 1);

  dim3 agrid(SEQ / 16, BATCH * NHEADS);        // 128 x 32 blocks, 1 wave each
  attn_kernel<<<agrid, 32, 0, stream>>>(Qh, Kh, Vt, aout);

  gemm_kernel<<<ggrid, 128, 0, stream>>>(aout, woT, bo, (void*)out, 2);
}
